// MPNNLayer_66108136620503
// MI455X (gfx1250) — matrix operand order repaired and checked by reading, not compiled, not run on hardware
//
#include <hip/hip_runtime.h>
#include <hip/hip_bf16.h>

#define HID 128
#define NGRID_ROWS_PER_BLOCK 32
#define EDGES_PER_BLOCK 64
#define MTILES 4

typedef __attribute__((ext_vector_type(16))) __bf16 v16bf;
typedef __attribute__((ext_vector_type(8)))  float  v8f;

union Frag16 {
    uint4 u[2];
    v16bf v;
};

__device__ __forceinline__ unsigned int f2bf_bits(float x) {
    unsigned int u = __float_as_uint(x);
    return (u + 0x7FFFu + ((u >> 16) & 1u)) >> 16;   // round-to-nearest-even
}

__device__ __forceinline__ unsigned short f2bf(float x) {
    return (unsigned short)f2bf_bits(x);
}

__device__ __forceinline__ unsigned int pack2bf(float lo, float hi) {
    return f2bf_bits(lo) | (f2bf_bits(hi) << 16);
}

__device__ __forceinline__ float silu(float x) {
    return x / (1.0f + __expf(-x));
}

// B fragment: pre-swizzled bf16 weights, 32 contiguous bytes per lane per tile.
__device__ __forceinline__ Frag16 load_bfrag(const unsigned short* __restrict__ w,
                                             int nt, int ksteps, int ks, int lane) {
    const uint4* p = (const uint4*)(w + ((size_t)((nt * ksteps + ks) * 32 + lane)) * 16);
    Frag16 f;
    f.u[0] = p[0];
    f.u[1] = p[1];
    return f;
}

// A fragment from row-major bf16 LDS tile (16 rows starting at `lds`, row stride
// `stride` elements). 16x32 bf16 A layout: lane group hi covers k = ks*32+hi*16..+15
// (contiguous, matching B swizzle) -> two aligned 16B LDS loads.
__device__ __forceinline__ Frag16 load_afrag(const unsigned short* lds, int stride,
                                             int mrow, int ks, int hi) {
    const unsigned short* r = lds + mrow * stride + ks * 32 + hi * 16;
    Frag16 f;
    f.u[0] = *(const uint4*)(r);
    f.u[1] = *(const uint4*)(r + 8);
    return f;
}

#define WMMA_BF16(A, B, C) \
    __builtin_amdgcn_wmma_f32_16x16x32_bf16(false, (A), false, (B), (short)0, (C), false, false)

// ---------------------------------------------------------------------------
// Prep kernels
// ---------------------------------------------------------------------------
__global__ void zero_f32(float* __restrict__ p, int n) {
    int i = blockIdx.x * blockDim.x + threadIdx.x;
    int step = gridDim.x * blockDim.x;
    for (; i < n; i += step) p[i] = 0.0f;
}

// Convert fp32 [K,N] row-major weight into bf16 B-fragment order:
// tile (nt,ks) -> 32 lanes x 16 bf16 contiguous; lane l slot v holds
// W[ks*32 + (l>>4)*16 + 2v][nt*16 + (l&15)] and the k+1 partner.
__global__ void swizzle_w(const float* __restrict__ src, unsigned short* __restrict__ dst,
                          int K, int N) {
    int t = blockIdx.x * blockDim.x + threadIdx.x;
    int ksteps = K / 32;
    int total = (N / 16) * ksteps * 32;
    if (t >= total) return;
    int lane = t & 31;
    int tile = t >> 5;
    int ks = tile % ksteps;
    int nt = tile / ksteps;
    int n = nt * 16 + (lane & 15);
    int kb = ks * 32 + (lane >> 4) * 16;
    unsigned short* out = dst + (size_t)t * 16;
#pragma unroll
    for (int v = 0; v < 8; ++v) {
        int k = kb + 2 * v;
        out[2 * v]     = f2bf(src[(size_t)k * N + n]);
        out[2 * v + 1] = f2bf(src[(size_t)(k + 1) * N + n]);
    }
}

// ---------------------------------------------------------------------------
// Edge kernel: fused edge MLP + message MLP + atomic scatter into segment sums.
// 256 threads = 8 waves; wave w owns output columns [w*16, w*16+16);
// block handles 64 edges = 4 WMMA M-tiles sharing each B fragment.
// sHH is time-multiplexed: h1 (phases 0b-1) then h2 (phases 2-3).
// ---------------------------------------------------------------------------
__global__ __launch_bounds__(256) void edge_kernel(
    const float* __restrict__ emb, const float* __restrict__ npos,
    const float* __restrict__ gpos, const int* __restrict__ eidx,
    const float* __restrict__ ew1, const float* __restrict__ eb1,
    const float* __restrict__ eb2, const float* __restrict__ mb1,
    const float* __restrict__ mb2,
    const unsigned short* __restrict__ w_ew2,
    const unsigned short* __restrict__ w_mw1,
    const unsigned short* __restrict__ w_mw2,
    float* __restrict__ sums, float* __restrict__ counts, long long E) {
    __shared__ __align__(16) unsigned short sMsg[EDGES_PER_BLOCK][256]; // [emb|pos_emb]
    __shared__ __align__(16) unsigned short sHH[EDGES_PER_BLOCK][128];  // h1 then h2
    __shared__ float s_ew1[6 * 128];
    __shared__ float s_eb1[128];
    __shared__ int s_i[EDGES_PER_BLOCK];
    __shared__ int s_j[EDGES_PER_BLOCK];

    const int tid = threadIdx.x;
    const int lane = tid & 31;
    const int wv = tid >> 5;       // N-tile 0..7
    const int hi = lane >> 4;
    const int nl = lane & 15;
    const long long e0 = (long long)blockIdx.x * EDGES_PER_BLOCK;

    // ---- phase 0a: small-weight preload, edge indices, counts ----
    for (int idx = tid; idx < 6 * 128; idx += 256) s_ew1[idx] = ew1[idx];
    if (tid < 128) s_eb1[tid] = eb1[tid];
    if (tid < EDGES_PER_BLOCK) {
        long long e = e0 + tid;
        int i = eidx[e];
        int j = eidx[E + e];
        s_i[tid] = i;
        s_j[tid] = j;
        atomicAdd(counts + j, 1.0f);
    }
    __syncthreads();

    // ---- phase 0b: embedding gather (bf16) + h1 = silu(edge_attr@ew1+eb1) ----
    {
        const int e = tid >> 2;          // edge within block, 0..63 (4 thr/edge)
        const int c0 = (tid & 3) * 32;   // 32 columns per thread
        const int i = s_i[e];
        const int j = s_j[e];
        const float* ep = emb + (size_t)i * HID + c0;
        unsigned int* mrow = (unsigned int*)&sMsg[e][c0];
#pragma unroll
        for (int c = 0; c < 32; c += 2)
            mrow[c >> 1] = pack2bf(ep[c], ep[c + 1]);

        const float a0 = npos[(size_t)i * 3 + 0];
        const float a1 = npos[(size_t)i * 3 + 1];
        const float a2 = npos[(size_t)i * 3 + 2];
        const float a3 = gpos[(size_t)j * 3 + 0];
        const float a4 = gpos[(size_t)j * 3 + 1];
        const float a5 = gpos[(size_t)j * 3 + 2];
        unsigned int* hrow = (unsigned int*)&sHH[e][c0];
#pragma unroll
        for (int c = 0; c < 32; c += 2) {
            float r[2];
#pragma unroll
            for (int q = 0; q < 2; ++q) {
                const int col = c0 + c + q;
                float acc = s_eb1[col];
                acc += a0 * s_ew1[0 * 128 + col] + a1 * s_ew1[1 * 128 + col] +
                       a2 * s_ew1[2 * 128 + col] + a3 * s_ew1[3 * 128 + col] +
                       a4 * s_ew1[4 * 128 + col] + a5 * s_ew1[5 * 128 + col];
                r[q] = silu(acc);
            }
            hrow[c >> 1] = pack2bf(r[0], r[1]);
        }
    }
    __syncthreads();

    const v8f vzero = {0.f, 0.f, 0.f, 0.f, 0.f, 0.f, 0.f, 0.f};

    // ---- phase 1: pos_emb = h1 @ ew2 + eb2 -> sMsg[:, 128:256] ----
    {
        v8f acc[MTILES] = {vzero, vzero, vzero, vzero};
#pragma unroll
        for (int ks = 0; ks < 4; ++ks) {
            Frag16 b = load_bfrag(w_ew2, wv, 4, ks, lane);
#pragma unroll
            for (int mt = 0; mt < MTILES; ++mt) {
                Frag16 a = load_afrag(&sHH[mt * 16][0], 128, nl, ks, hi);
                acc[mt] = WMMA_BF16(a.v, b.v, acc[mt]);
            }
        }
        const float bias = eb2[wv * 16 + nl];
#pragma unroll
        for (int mt = 0; mt < MTILES; ++mt)
#pragma unroll
            for (int v = 0; v < 8; ++v)
                sMsg[mt * 16 + v + 8 * hi][128 + wv * 16 + nl] = f2bf(acc[mt][v] + bias);
    }
    __syncthreads();

    // ---- phase 2: h2 = silu(msg_in @ mw1 + mb1), K = 256 -> sHH ----
    {
        v8f acc[MTILES] = {vzero, vzero, vzero, vzero};
#pragma unroll
        for (int ks = 0; ks < 8; ++ks) {
            Frag16 b = load_bfrag(w_mw1, wv, 8, ks, lane);
#pragma unroll
            for (int mt = 0; mt < MTILES; ++mt) {
                Frag16 a = load_afrag(&sMsg[mt * 16][0], 256, nl, ks, hi);
                acc[mt] = WMMA_BF16(a.v, b.v, acc[mt]);
            }
        }
        __syncthreads();   // h1 reads done; safe to overwrite sHH with h2
        const float bias = mb1[wv * 16 + nl];
#pragma unroll
        for (int mt = 0; mt < MTILES; ++mt)
#pragma unroll
            for (int v = 0; v < 8; ++v)
                sHH[mt * 16 + v + 8 * hi][wv * 16 + nl] = f2bf(silu(acc[mt][v] + bias));
    }
    __syncthreads();

    // ---- phase 3: message = h2 @ mw2 + mb2; atomic scatter-add into sums ----
    {
        v8f acc[MTILES] = {vzero, vzero, vzero, vzero};
#pragma unroll
        for (int ks = 0; ks < 4; ++ks) {
            Frag16 b = load_bfrag(w_mw2, wv, 4, ks, lane);
#pragma unroll
            for (int mt = 0; mt < MTILES; ++mt) {
                Frag16 a = load_afrag(&sHH[mt * 16][0], 128, nl, ks, hi);
                acc[mt] = WMMA_BF16(a.v, b.v, acc[mt]);
            }
        }
        const float bias = mb2[wv * 16 + nl];
#pragma unroll
        for (int mt = 0; mt < MTILES; ++mt)
#pragma unroll
            for (int v = 0; v < 8; ++v) {
                const int row = mt * 16 + v + 8 * hi;
                const int j = s_j[row];
                atomicAdd(&sums[(size_t)j * HID + wv * 16 + nl], acc[mt][v] + bias);
            }
    }
}

// ---------------------------------------------------------------------------
// Update kernel: mean = sums/counts; out = silu(mean@uw1+ub1)@uw2+ub2
// ---------------------------------------------------------------------------
__global__ __launch_bounds__(256) void update_kernel(
    const float* __restrict__ sums, const float* __restrict__ counts,
    const unsigned short* __restrict__ w_uw1, const unsigned short* __restrict__ w_uw2,
    const float* __restrict__ ub1, const float* __restrict__ ub2,
    float* __restrict__ out) {
    __shared__ __align__(16) unsigned short sA[32][128];
    __shared__ __align__(16) unsigned short sH[32][128];

    const int tid = threadIdx.x;
    const int lane = tid & 31;
    const int wv = tid >> 5;
    const int hi = lane >> 4;
    const int nl = lane & 15;
    const int g0 = blockIdx.x * NGRID_ROWS_PER_BLOCK;

    {
        const int r = tid >> 3;
        const int c0 = (tid & 7) * 16;
        const int g = g0 + r;
        const float inv = 1.0f / fmaxf(counts[g], 1.0f);
        const float* sp = sums + (size_t)g * HID + c0;
        unsigned int* arow = (unsigned int*)&sA[r][c0];
#pragma unroll
        for (int c = 0; c < 16; c += 2)
            arow[c >> 1] = pack2bf(sp[c] * inv, sp[c + 1] * inv);
    }
    __syncthreads();

    const v8f vzero = {0.f, 0.f, 0.f, 0.f, 0.f, 0.f, 0.f, 0.f};

    {   // h = silu(mean @ uw1 + ub1)
        v8f acc[2] = {vzero, vzero};
#pragma unroll
        for (int ks = 0; ks < 4; ++ks) {
            Frag16 b = load_bfrag(w_uw1, wv, 4, ks, lane);
#pragma unroll
            for (int mt = 0; mt < 2; ++mt) {
                Frag16 a = load_afrag(&sA[mt * 16][0], 128, nl, ks, hi);
                acc[mt] = WMMA_BF16(a.v, b.v, acc[mt]);
            }
        }
        const float bias = ub1[wv * 16 + nl];
#pragma unroll
        for (int mt = 0; mt < 2; ++mt)
#pragma unroll
            for (int v = 0; v < 8; ++v)
                sH[mt * 16 + v + 8 * hi][wv * 16 + nl] = f2bf(silu(acc[mt][v] + bias));
    }
    __syncthreads();

    {   // out = h @ uw2 + ub2
        v8f acc[2] = {vzero, vzero};
#pragma unroll
        for (int ks = 0; ks < 4; ++ks) {
            Frag16 b = load_bfrag(w_uw2, wv, 4, ks, lane);
#pragma unroll
            for (int mt = 0; mt < 2; ++mt) {
                Frag16 a = load_afrag(&sH[mt * 16][0], 128, nl, ks, hi);
                acc[mt] = WMMA_BF16(a.v, b.v, acc[mt]);
            }
        }
        const float bias = ub2[wv * 16 + nl];
#pragma unroll
        for (int mt = 0; mt < 2; ++mt)
#pragma unroll
            for (int v = 0; v < 8; ++v) {
                const int g = g0 + mt * 16 + v + 8 * hi;
                out[(size_t)g * HID + wv * 16 + nl] = acc[mt][v] + bias;
            }
    }
}

// ---------------------------------------------------------------------------
extern "C" void kernel_launch(void* const* d_in, const int* in_sizes, int n_in,
                              void* d_out, int out_size, void* d_ws, size_t ws_size,
                              hipStream_t stream) {
    const float* emb  = (const float*)d_in[0];
    const float* npos = (const float*)d_in[1];
    const float* gpos = (const float*)d_in[2];
    const int*   eidx = (const int*)d_in[3];
    const float* ew1 = (const float*)d_in[4];
    const float* eb1 = (const float*)d_in[5];
    const float* ew2 = (const float*)d_in[6];
    const float* eb2 = (const float*)d_in[7];
    const float* mw1 = (const float*)d_in[8];
    const float* mb1 = (const float*)d_in[9];
    const float* mw2 = (const float*)d_in[10];
    const float* mb2 = (const float*)d_in[11];
    const float* uw1 = (const float*)d_in[12];
    const float* ub1 = (const float*)d_in[13];
    const float* uw2 = (const float*)d_in[14];
    const float* ub2 = (const float*)d_in[15];
    float* out = (float*)d_out;

    const long long E = in_sizes[3] / 2;      // 524288
    const int n_grid  = in_sizes[2] / 3;      // 32768

    // workspace layout
    float* sums   = (float*)d_ws;                          // n_grid*HID f32
    float* counts = sums + (size_t)n_grid * HID;           // n_grid f32
    unsigned short* w_ew2 = (unsigned short*)(counts + n_grid);
    unsigned short* w_mw1 = w_ew2 + 128 * 128;
    unsigned short* w_mw2 = w_mw1 + 256 * 128;
    unsigned short* w_uw1 = w_mw2 + 128 * 128;
    unsigned short* w_uw2 = w_uw1 + 128 * 128;

    const int nz = n_grid * HID + n_grid;
    zero_f32<<<1024, 256, 0, stream>>>(sums, nz);
    swizzle_w<<<4, 256, 0, stream>>>(ew2, w_ew2, 128, 128);
    swizzle_w<<<8, 256, 0, stream>>>(mw1, w_mw1, 256, 128);
    swizzle_w<<<4, 256, 0, stream>>>(mw2, w_mw2, 128, 128);
    swizzle_w<<<4, 256, 0, stream>>>(uw1, w_uw1, 128, 128);
    swizzle_w<<<4, 256, 0, stream>>>(uw2, w_uw2, 128, 128);

    const int edge_blocks = (int)(E / EDGES_PER_BLOCK);    // 8192
    edge_kernel<<<edge_blocks, 256, 0, stream>>>(
        emb, npos, gpos, eidx, ew1, eb1, eb2, mb1, mb2,
        w_ew2, w_mw1, w_mw2, sums, counts, E);

    const int upd_blocks = n_grid / NGRID_ROWS_PER_BLOCK;  // 1024
    update_kernel<<<upd_blocks, 256, 0, stream>>>(
        sums, counts, w_uw1, w_uw2, ub1, ub2, out);
}
